// VectorGNN_49864570306577
// MI455X (gfx1250) — matrix operand order repaired and checked by reading, not compiled, not run on hardware
//
#include <hip/hip_runtime.h>
#include <hip/hip_bf16.h>

// ------------------------------------------------------------------
// VectorGNN forward on gfx1250 (CDNA5). All GEMMs use
// v_wmma_f32_16x16x32_bf16 (bf16 operands, f32 accumulate); residual
// stream stays fp32. Edge scatter uses global_atomic_add_f32.
// B fragments live in LDS pre-swizzled (32B/lane -> 2x ds_load_b128);
// A fragments load as two contiguous 8-element runs (global_load_b128).
// k_mlp waves own 32 rows (2 M-tiles) so each B fragment feeds 2 WMMAs.
// Activation math uses v_exp_f32 + v_rcp_f32 (no IEEE div sequences).
// ------------------------------------------------------------------

#define NN      50000      // N_NODES
#define NE      800000     // N_EDGES
#define ND      128        // NODE_DIM
#define MD      32         // MSG_DIM
#define NR      32         // NUM_RADIAL
#define CUTOFF  5.0f
#define PI_F    3.14159265358979323846f

typedef __attribute__((ext_vector_type(16))) __bf16 bf16x16;
typedef __attribute__((ext_vector_type(8)))  __bf16 bf16x8;
typedef __attribute__((ext_vector_type(8)))  float  floatx8;
typedef __attribute__((ext_vector_type(4)))  float  floatx4;

// ---------------- WMMA helpers (layouts per CDNA5 ISA) -------
// A (16-bit, 16x32): lane&15 = M; per lane, elements e=0..7 -> k=kb+e,
//   e=8..15 -> k=kb+16+(e-8), kb = (lane>=16)?8:0  (two contiguous runs).
// B (16-bit, 32x16): lane&15 = N; same k order as A (kb + e, kb + 16 + (e-8)).
// C/D f32 16x16: lane col = lane&15, rows r + 8*(lane>=16).

__device__ __forceinline__ bf16x16 combine8(bf16x8 lo, bf16x8 hi) {
  return __builtin_shufflevector(lo, hi, 0, 1, 2, 3, 4, 5, 6, 7,
                                 8, 9, 10, 11, 12, 13, 14, 15);
}

// B fragment from swizzled LDS: lane's 16 bf16 contiguous (32 bytes).
__device__ __forceinline__ bf16x16 load_b_sw(const __bf16* sw, int tile, int lane) {
  const bf16x8* q = (const bf16x8*)(sw + ((tile * 32 + lane) << 4));
  return combine8(q[0], q[1]);
}

// A fragment from global f32 row (two 8-float runs -> 4x b128 loads + cvt).
__device__ __forceinline__ bf16x16 load_a_g32(const float* row, int lane) {
  int kb = (lane >= 16) ? 8 : 0;
  floatx4 f0 = *(const floatx4*)(row + kb);
  floatx4 f1 = *(const floatx4*)(row + kb + 4);
  floatx4 f2 = *(const floatx4*)(row + kb + 16);
  floatx4 f3 = *(const floatx4*)(row + kb + 20);
  bf16x16 a;
  a[0]  = (__bf16)f0[0]; a[1]  = (__bf16)f0[1]; a[2]  = (__bf16)f0[2]; a[3]  = (__bf16)f0[3];
  a[4]  = (__bf16)f1[0]; a[5]  = (__bf16)f1[1]; a[6]  = (__bf16)f1[2]; a[7]  = (__bf16)f1[3];
  a[8]  = (__bf16)f2[0]; a[9]  = (__bf16)f2[1]; a[10] = (__bf16)f2[2]; a[11] = (__bf16)f2[3];
  a[12] = (__bf16)f3[0]; a[13] = (__bf16)f3[1]; a[14] = (__bf16)f3[2]; a[15] = (__bf16)f3[3];
  return a;
}

// A fragment from row-major bf16 LDS row (two 16B runs -> 2x ds_load_b128).
__device__ __forceinline__ bf16x16 load_a_lds16(const __bf16* row, int lane) {
  int kb = (lane >= 16) ? 8 : 0;
  bf16x8 lo = *(const bf16x8*)(row + kb);
  bf16x8 hi = *(const bf16x8*)(row + kb + 16);
  return combine8(lo, hi);
}

__device__ __forceinline__ floatx8 wmma_bf16(bf16x16 a, bf16x16 b, floatx8 c) {
  return __builtin_amdgcn_wmma_f32_16x16x32_bf16(false, a, false, b, (short)0, c, false, false);
}

// Stage W [K x ldn] (f32, K = KT*32, N = NT*16) into fragment-swizzled bf16:
//   dst[((kt*NT+nt)*32 + lane)*16 + e] = W[(kt*32 + kb8 + e + (e>=8?8:0))*ldn + nt*16 + (lane&15)]
__device__ __forceinline__ void stage_b_sw(const float* __restrict__ W, __bf16* dst,
                                           int KT, int NT, int ldn,
                                           int tid, int nthreads) {
  int total = KT * NT * 512;
  for (int t = tid; t < total; t += nthreads) {
    int e = t & 15, ln = (t >> 4) & 31, tile = t >> 9;
    int kt = tile / NT, nt = tile - kt * NT;
    int k = kt * 32 + ((ln >= 16) ? 8 : 0) + e + ((e >= 8) ? 8 : 0);
    int n = nt * 16 + (ln & 15);
    dst[t] = (__bf16)W[k * ldn + n];
  }
}

// ---------------- math helpers -----------------------------------------

__device__ __forceinline__ float fast_rcp(float x) { return __builtin_amdgcn_rcpf(x); }

// branchless tanh: sign(x) * (1 - e) * rcp(1 + e), e = exp(-2|x|)
__device__ __forceinline__ float tanh_fast(float x) {
  float ax = fabsf(x);
  float e  = __expf(-2.0f * ax);
  float t  = (1.0f - e) * fast_rcp(1.0f + e);
  return copysignf(t, x);
}

__device__ __forceinline__ float gelu_f(float x) {
  const float c0 = 0.7978845608028654f, c1 = 0.044715f;
  float t = tanh_fast(c0 * (x + c1 * x * x * x));
  return 0.5f * x * (1.0f + t);
}
__device__ __forceinline__ float sigmoid_f(float x) {
  return fast_rcp(1.0f + __expf(-x));
}

// DimeNet envelope, p = 6: env = 1/d - 28 d^5 + 48 d^6 - 21 d^7  (0 for d>=1)
__device__ __forceinline__ float envelope_f(float d) {
  if (d >= 1.0f) return 0.0f;
  float d2 = d * d, d4 = d2 * d2, d5 = d4 * d;
  return fast_rcp(d) + d5 * (-28.0f + d * (48.0f - 21.0f * d));
}

__device__ __forceinline__ void atomAddF(float* p, float v) {
  unsafeAtomicAdd(p, v);   // -> global_atomic_add_f32
}

// ---------------- embedding --------------------------------------------

__global__ void __launch_bounds__(256, 1)
k_embed(const float* __restrict__ x, const float* __restrict__ atom_emb,
        const float* __restrict__ emb_W, const float* __restrict__ emb_b,
        float* __restrict__ h) {
  int idx = blockIdx.x * blockDim.x + threadIdx.x;
  if (idx >= NN * ND) return;
  int n = idx >> 7, d = idx & 127;
  int atom = (int)x[n * 6];
  atom = atom < 0 ? 0 : (atom > 100 ? 100 : atom);
  float acc = emb_b[d];
#pragma unroll
  for (int k = 0; k < 5; ++k) acc += atom_emb[atom * 5 + k] * emb_W[k * ND + d];
#pragma unroll
  for (int k = 0; k < 5; ++k) acc += x[n * 6 + 1 + k] * emb_W[(5 + k) * ND + d];
  h[idx] = acc;
}

__global__ void __launch_bounds__(256, 1)
k_vinit(const float* __restrict__ v, const float* __restrict__ vec_W,
        float* __restrict__ vv) {
  int idx = blockIdx.x * blockDim.x + threadIdx.x;
  if (idx >= 3 * NN * ND) return;
  int r = idx >> 7, d = idx & 127;
  vv[idx] = v[r] * vec_W[d];
}

// ---------------- fused vector_mlp layer -------------------------------
// h += gelu(h@Ws+bs); gate = sigmoid(h@Wg); vv += (vv@Wv)*gate
// block = 128 threads = 4 waves; each wave owns 32 rows (two 16-row tiles)
// so every B fragment loaded from LDS feeds two WMMAs.

__global__ void __launch_bounds__(128, 1)
k_mlp(float* __restrict__ h, float* __restrict__ vv,
      const float* __restrict__ Ws, const float* __restrict__ bs,
      const float* __restrict__ Wg, const float* __restrict__ Wv) {
  extern __shared__ char smem[];
  __bf16* sWs = (__bf16*)smem;           // 16384 (swizzled)
  __bf16* sWg = sWs + ND * ND;
  __bf16* sWv = sWg + ND * ND;
  __bf16* sH  = sWv + ND * ND;           // 4 waves * 32*128 bf16 (row-major)
  __bf16* sG  = sH + 4 * 4096;           // 4 waves * 32*128 bf16

  int tid = threadIdx.x;
  stage_b_sw(Ws, sWs, 4, 8, ND, tid, 128);
  stage_b_sw(Wg, sWg, 4, 8, ND, tid, 128);
  stage_b_sw(Wv, sWv, 4, 8, ND, tid, 128);
  __syncthreads();

  int wave = tid >> 5, lane = tid & 31;
  int row0 = (blockIdx.x * 4 + wave) * 32;
  int m    = lane & 15;
  int rA = row0 + m;      if (rA > NN - 1) rA = NN - 1;
  int rB = row0 + 16 + m; if (rB > NN - 1) rB = NN - 1;
  int col0 = lane & 15, rowoff = (lane >= 16) ? 8 : 0;
  __bf16* myH = sH + wave * 4096;
  __bf16* myG = sG + wave * 4096;

  // ---- GEMM1: t = h @ Ws ; h += gelu(t + bs) --------------------------
  bf16x16 A0[4], A1[4];
#pragma unroll
  for (int kt = 0; kt < 4; ++kt) {
    A0[kt] = load_a_g32(h + rA * ND + kt * 32, lane);
    A1[kt] = load_a_g32(h + rB * ND + kt * 32, lane);
  }

  for (int nt = 0; nt < 8; ++nt) {
    bf16x16 B[4];
#pragma unroll
    for (int kt = 0; kt < 4; ++kt) B[kt] = load_b_sw(sWs, kt * 8 + nt, lane);
    floatx8 acc0 = {0.f, 0.f, 0.f, 0.f, 0.f, 0.f, 0.f, 0.f};
    floatx8 acc1 = {0.f, 0.f, 0.f, 0.f, 0.f, 0.f, 0.f, 0.f};
#pragma unroll
    for (int kt = 0; kt < 4; ++kt) acc0 = wmma_bf16(A0[kt], B[kt], acc0);
#pragma unroll
    for (int kt = 0; kt < 4; ++kt) acc1 = wmma_bf16(A1[kt], B[kt], acc1);

    int n = nt * 16 + col0;
    float bsn = bs[n];
#pragma unroll
    for (int half = 0; half < 2; ++half) {
      floatx8 acc = half ? acc1 : acc0;
      int rbase = row0 + half * 16 + rowoff;
#pragma unroll
      for (int r = 0; r < 8; ++r) {
        int row = rbase + r;
        int rc  = row < NN ? row : NN - 1;
        float hnew = h[rc * ND + n] + gelu_f(acc[r] + bsn);
        if (row < NN) h[row * ND + n] = hnew;
        myH[(half * 16 + rowoff + r) * ND + n] = (__bf16)hnew;
      }
    }
  }
  __syncthreads();

  // ---- GEMM2: gate = sigmoid(h_new @ Wg) ------------------------------
  bf16x16 A20[4], A21[4];
#pragma unroll
  for (int kt = 0; kt < 4; ++kt) {
    A20[kt] = load_a_lds16(myH + m * ND + kt * 32, lane);
    A21[kt] = load_a_lds16(myH + (16 + m) * ND + kt * 32, lane);
  }

  for (int nt = 0; nt < 8; ++nt) {
    bf16x16 B[4];
#pragma unroll
    for (int kt = 0; kt < 4; ++kt) B[kt] = load_b_sw(sWg, kt * 8 + nt, lane);
    floatx8 acc0 = {0.f, 0.f, 0.f, 0.f, 0.f, 0.f, 0.f, 0.f};
    floatx8 acc1 = {0.f, 0.f, 0.f, 0.f, 0.f, 0.f, 0.f, 0.f};
#pragma unroll
    for (int kt = 0; kt < 4; ++kt) acc0 = wmma_bf16(A20[kt], B[kt], acc0);
#pragma unroll
    for (int kt = 0; kt < 4; ++kt) acc1 = wmma_bf16(A21[kt], B[kt], acc1);

    int n = nt * 16 + col0;
#pragma unroll
    for (int r = 0; r < 8; ++r) {
      myG[(rowoff + r) * ND + n]      = (__bf16)sigmoid_f(acc0[r]);
      myG[(16 + rowoff + r) * ND + n] = (__bf16)sigmoid_f(acc1[r]);
    }
  }
  __syncthreads();

  // ---- vector track: vv += (vv @ Wv) * gate ---------------------------
  for (int c = 0; c < 3; ++c) {
    bf16x16 Av0[4], Av1[4];
#pragma unroll
    for (int kt = 0; kt < 4; ++kt) {
      Av0[kt] = load_a_g32(vv + (rA * 3 + c) * ND + kt * 32, lane);
      Av1[kt] = load_a_g32(vv + (rB * 3 + c) * ND + kt * 32, lane);
    }

    for (int nt = 0; nt < 8; ++nt) {
      bf16x16 B[4];
#pragma unroll
      for (int kt = 0; kt < 4; ++kt) B[kt] = load_b_sw(sWv, kt * 8 + nt, lane);
      floatx8 acc0 = {0.f, 0.f, 0.f, 0.f, 0.f, 0.f, 0.f, 0.f};
      floatx8 acc1 = {0.f, 0.f, 0.f, 0.f, 0.f, 0.f, 0.f, 0.f};
#pragma unroll
      for (int kt = 0; kt < 4; ++kt) acc0 = wmma_bf16(Av0[kt], B[kt], acc0);
#pragma unroll
      for (int kt = 0; kt < 4; ++kt) acc1 = wmma_bf16(Av1[kt], B[kt], acc1);

      int n = nt * 16 + col0;
#pragma unroll
      for (int half = 0; half < 2; ++half) {
        floatx8 acc = half ? acc1 : acc0;
#pragma unroll
        for (int r = 0; r < 8; ++r) {
          int row = row0 + half * 16 + rowoff + r;
          if (row < NN) {
            int vi = (row * 3 + c) * ND + n;
            float g = (float)myG[(half * 16 + rowoff + r) * ND + n];
            vv[vi] = vv[vi] + acc[r] * g;
          }
        }
      }
    }
  }
}

// ---------------- projection [nrows,128] @ [128,32] --------------------
// out = (gelu?)(in @ W + b). 256 threads = 8 waves, 16 rows/wave.

__global__ void __launch_bounds__(256, 1)
k_proj(const float* __restrict__ in, int nrows,
       const float* __restrict__ W, const float* __restrict__ b,
       int do_gelu, float* __restrict__ out) {
  extern __shared__ char smem[];
  __bf16* sW = (__bf16*)smem; // 128*32 swizzled (KT=4, NT=2)
  int tid = threadIdx.x;
  stage_b_sw(W, sW, 4, 2, MD, tid, 256);
  __syncthreads();

  int wave = tid >> 5, lane = tid & 31;
  int row0  = (blockIdx.x * 8 + wave) * 16;
  int m     = lane & 15;
  int rload = row0 + m; if (rload > nrows - 1) rload = nrows - 1;
  int col0  = lane & 15, rowoff = (lane >= 16) ? 8 : 0;

  bf16x16 A[4];
#pragma unroll
  for (int kt = 0; kt < 4; ++kt)
    A[kt] = load_a_g32(in + rload * ND + kt * 32, lane);

  for (int nt = 0; nt < 2; ++nt) {
    bf16x16 B[4];
#pragma unroll
    for (int kt = 0; kt < 4; ++kt) B[kt] = load_b_sw(sW, kt * 2 + nt, lane);
    floatx8 acc = {0.f, 0.f, 0.f, 0.f, 0.f, 0.f, 0.f, 0.f};
#pragma unroll
    for (int kt = 0; kt < 4; ++kt) acc = wmma_bf16(A[kt], B[kt], acc);
    int n  = nt * 16 + col0;
    float bn = b ? b[n] : 0.0f;
#pragma unroll
    for (int r = 0; r < 8; ++r) {
      int row = row0 + rowoff + r;
      if (row < nrows) {
        float t = acc[r] + bn;
        out[row * MD + n] = do_gelu ? gelu_f(t) : t;
      }
    }
  }
}

// ---------------- update [nrows,32] @ [32,128], residual add -----------

__global__ void __launch_bounds__(256, 1)
k_update(const float* __restrict__ agg, int nrows,
         const float* __restrict__ W, const float* __restrict__ b,
         float* __restrict__ out) {
  extern __shared__ char smem[];
  __bf16* sW = (__bf16*)smem; // 32*128 swizzled (KT=1, NT=8)
  int tid = threadIdx.x;
  stage_b_sw(W, sW, 1, 8, ND, tid, 256);
  __syncthreads();

  int wave = tid >> 5, lane = tid & 31;
  int row0  = (blockIdx.x * 8 + wave) * 16;
  int m     = lane & 15;
  int rload = row0 + m; if (rload > nrows - 1) rload = nrows - 1;
  int col0  = lane & 15, rowoff = (lane >= 16) ? 8 : 0;

  bf16x16 A = load_a_g32(agg + rload * MD, lane);

  for (int nt = 0; nt < 8; ++nt) {
    floatx8 acc = {0.f, 0.f, 0.f, 0.f, 0.f, 0.f, 0.f, 0.f};
    acc = wmma_bf16(A, load_b_sw(sW, nt, lane), acc);
    int n  = nt * 16 + col0;
    float bn = b ? b[n] : 0.0f;
#pragma unroll
    for (int r = 0; r < 8; ++r) {
      int row = row0 + rowoff + r;
      if (row < nrows) out[row * ND + n] += acc[r] + bn;
    }
  }
}

// ---------------- per-edge: rbf gates + gather/scatter -----------------
// 256 threads = 8 waves, 16 edges/wave. RBF recomputed on the fly.

__global__ void __launch_bounds__(256, 1)
k_edges(const int* __restrict__ ei, const float* __restrict__ pos,
        const float* __restrict__ msrc, const float* __restrict__ vmsrc,
        const float* __restrict__ We, const float* __restrict__ be,
        const float* __restrict__ vWe,
        float* __restrict__ agg_s, float* __restrict__ agg_v) {
  extern __shared__ char smem[];
  __bf16* sWe = (__bf16*)smem;   // 32*32 swizzled (KT=1, NT=2)
  __bf16* sVe = sWe + NR * MD;   // 32*32 swizzled
  __bf16* sR  = sVe + NR * MD;   // 8 waves * 16*32 row-major bf16

  int tid = threadIdx.x;
  stage_b_sw(We, sWe, 1, 2, MD, tid, 256);
  stage_b_sw(vWe, sVe, 1, 2, MD, tid, 256);

  int wave = tid >> 5, lane = tid & 31;
  int ebase = (blockIdx.x * 8 + wave) * 16;
  int el    = lane & 15, half = lane >> 4;
  int e     = ebase + el;
  int ec    = e < NE ? e : NE - 1;
  __builtin_prefetch(ei + ec + 128, 0, 0);

  int ii = ei[ec], jj = ei[NE + ec];
  float dx = pos[ii * 3 + 0] - pos[jj * 3 + 0];
  float dy = pos[ii * 3 + 1] - pos[jj * 3 + 1];
  float dz = pos[ii * 3 + 2] - pos[jj * 3 + 2];
  float d  = sqrtf(dx * dx + dy * dy + dz * dz) * (1.0f / CUTOFF);
  float env = envelope_f(d);

  __bf16* myR = sR + wave * 512;
#pragma unroll
  for (int k = 0; k < 16; ++k) {
    int kk = half * 16 + k;
    myR[el * NR + kk] = (__bf16)(env * __sinf((float)(kk + 1) * PI_F * d));
  }
  __syncthreads();

  bf16x16 A = load_a_lds16(myR + (lane & 15) * NR, lane);

  int col0 = lane & 15, rowoff = (lane >= 16) ? 8 : 0;
  for (int nt = 0; nt < 2; ++nt) {
    floatx8 z = {0.f, 0.f, 0.f, 0.f, 0.f, 0.f, 0.f, 0.f};
    floatx8 accS = wmma_bf16(A, load_b_sw(sWe, nt, lane), z);
    floatx8 accV = wmma_bf16(A, load_b_sw(sVe, nt, lane), z);
    int col = nt * 16 + col0;
    float bec = be[col];
#pragma unroll
    for (int r = 0; r < 8; ++r) {
      int ee = ebase + rowoff + r;
      if (ee < NE) {
        int si = ei[ee], sj = ei[NE + ee];
        float gs = accS[r] + bec;
        float gv = accV[r];
        atomAddF(&agg_s[si * MD + col], msrc[sj * MD + col] * gs);
#pragma unroll
        for (int c = 0; c < 3; ++c)
          atomAddF(&agg_v[(si * 3 + c) * MD + col],
                   vmsrc[(sj * 3 + c) * MD + col] * gv);
      }
    }
  }
}

// ---------------- host side --------------------------------------------
// Leaf order assumption: harness flattens setup_inputs() with JAX pytree
// semantics (dict keys sorted at every level):
//  0 batch, 1 edge_index, 2 params.atom_emb,
//  3..29 params.convs[k] x {We,Wm,Wu,be,bm,bu,vWe,vWm,vWu} (9 each),
//  30 emb_W, 31 emb_b,
//  32..55 fc[k][l] x {Wg,Ws,Wv,bs}, 56..63 fc0[l] x {Wg,Ws,Wv,bs},
//  64 freqs, 65 vec_emb_W, 66 pos, 67 v, 68 x.

extern "C" void kernel_launch(void* const* d_in, const int* in_sizes, int n_in,
                              void* d_out, int out_size, void* d_ws, size_t ws_size,
                              hipStream_t stream) {
  if (n_in < 69) return;

  const int*   edge_idx = (const int*)d_in[1];
  const float* atom_emb = (const float*)d_in[2];

  struct ConvP { const float *We, *Wm, *Wu, *be, *bm, *bu, *vWe, *vWm, *vWu; };
  ConvP convs[3];
  for (int k = 0; k < 3; ++k) {
    int b = 3 + 9 * k;
    convs[k].We  = (const float*)d_in[b + 0];
    convs[k].Wm  = (const float*)d_in[b + 1];
    convs[k].Wu  = (const float*)d_in[b + 2];
    convs[k].be  = (const float*)d_in[b + 3];
    convs[k].bm  = (const float*)d_in[b + 4];
    convs[k].bu  = (const float*)d_in[b + 5];
    convs[k].vWe = (const float*)d_in[b + 6];
    convs[k].vWm = (const float*)d_in[b + 7];
    convs[k].vWu = (const float*)d_in[b + 8];
  }
  const float* emb_W = (const float*)d_in[30];
  const float* emb_b = (const float*)d_in[31];
  const float* vec_W = (const float*)d_in[65];
  const float* pos   = (const float*)d_in[66];
  const float* v     = (const float*)d_in[67];
  const float* x     = (const float*)d_in[68];

  float* h  = (float*)d_out;
  float* vv = h + (size_t)NN * ND;

  float* wsf    = (float*)d_ws;
  float* m_buf  = wsf;                              // NN*32
  float* vm_buf = m_buf + (size_t)NN * MD;          // 3*NN*32
  float* agg_s  = vm_buf + (size_t)3 * NN * MD;     // NN*32
  float* agg_v  = agg_s + (size_t)NN * MD;          // 3*NN*32

  {
    int tot = NN * ND;
    k_embed<<<(tot + 255) / 256, 256, 0, stream>>>(x, atom_emb, emb_W, emb_b, h);
  }
  {
    int tot = 3 * NN * ND;
    k_vinit<<<(tot + 255) / 256, 256, 0, stream>>>(v, vec_W, vv);
  }

  const size_t MLP_LDS  = (size_t)(3 * ND * ND + 2 * 4 * 4096) * 2; // 163840
  const size_t PROJ_LDS = (size_t)ND * MD * 2;   // 8192
  const size_t UPD_LDS  = (size_t)MD * ND * 2;   // 8192
  const size_t EDGE_LDS = (size_t)(2 * NR * MD + 8 * 512) * 2; // 12288
  int mlp_blocks  = (NN + 127) / 128;
  int projN_blk   = (NN + 127) / 128;
  int projV_blk   = (3 * NN + 127) / 128;
  int edge_blk    = (NE + 127) / 128;

  auto launch_mlp = [&](int base, int l) {
    const float* Wg  = (const float*)d_in[base + l * 4 + 0];
    const float* Wsp = (const float*)d_in[base + l * 4 + 1];
    const float* Wv  = (const float*)d_in[base + l * 4 + 2];
    const float* bsp = (const float*)d_in[base + l * 4 + 3];
    k_mlp<<<mlp_blocks, 128, MLP_LDS, stream>>>(h, vv, Wsp, bsp, Wg, Wv);
  };

  // fc0
  launch_mlp(56, 0);
  launch_mlp(56, 1);

  for (int k = 0; k < 3; ++k) {
    const ConvP& c = convs[k];
    k_proj<<<projN_blk, 256, PROJ_LDS, stream>>>(h, NN, c.Wm, c.bm, 1, m_buf);
    k_proj<<<projV_blk, 256, PROJ_LDS, stream>>>(vv, 3 * NN, c.vWm, nullptr, 0, vm_buf);

    hipMemsetAsync(agg_s, 0, (size_t)NN * MD * sizeof(float), stream);
    hipMemsetAsync(agg_v, 0, (size_t)3 * NN * MD * sizeof(float), stream);

    k_edges<<<edge_blk, 256, EDGE_LDS, stream>>>(edge_idx, pos, m_buf, vm_buf,
                                                 c.We, c.be, c.vWe, agg_s, agg_v);

    k_update<<<projN_blk, 256, UPD_LDS, stream>>>(agg_s, NN, c.Wu, c.bu, h);
    k_update<<<projV_blk, 256, UPD_LDS, stream>>>(agg_v, 3 * NN, c.vWu, nullptr, vv);

    launch_mlp(32 + k * 8, 0);
    launch_mlp(32 + k * 8, 1);
  }
}
